// MS_Attention_linear_61667140436268
// MI455X (gfx1250) — compile-verified, matrix operand order checked by reading, not run on hardware
//
#include <hip/hip_runtime.h>

typedef __attribute__((ext_vector_type(16))) _Float16 v16h;
typedef __attribute__((ext_vector_type(8)))  _Float16 v8h;
typedef __attribute__((ext_vector_type(8)))  float    v8f;
typedef __attribute__((ext_vector_type(4)))  float    v4f;

#define TB_TOTAL 128
#define C_DIM    512
#define N_POS    256
#define HEADS    8
#define HD       64
#define QKV_LD   1536

// ---------------------------------------------------------------------------
// WMMA helpers (CDNA5 wave32, V_WMMA_F32_16X16X32_F16)
//
// 16-bit A-operand layout (ISA 7.12.2): lane = (halfId<<4)|r holds row r,
// K in [halfId*8, halfId*8+8) and [16+halfId*8, ...). B-operand (K x N) uses
// the same per-lane layout over a [N][K] row-major tile (lane's column = r).
// So both operands load as two 16-byte LDS reads from an [outdim][k] tile.
// ---------------------------------------------------------------------------
__device__ __forceinline__ v16h load_frag16(const _Float16* __restrict__ base, int stride)
{
  const int lane = threadIdx.x & 31;
  const int r    = lane & 15;
  const int koff = (lane >> 4) << 3;   // 0 or 8
  const _Float16* p = base + (size_t)r * stride + koff;
  v8h lo = *(const v8h*)(p);
  v8h hi = *(const v8h*)(p + 16);
  v16h f;
#pragma unroll
  for (int i = 0; i < 8; ++i) { f[i] = lo[i]; f[i + 8] = hi[i]; }
  return f;
}

__device__ __forceinline__ v8f wmma_f16(v16h a, v16h b, v8f c)
{
  return __builtin_amdgcn_wmma_f32_16x16x32_f16(false, a, false, b, (short)0, c, false, false);
}

// q_if(x)/D_NORM : round-to-nearest-even of clamp(x,0,8), then /8
__device__ __forceinline__ float qspike(float v)
{
  v = fminf(fmaxf(v, 0.0f), 8.0f);
  return rintf(v) * 0.125f;
}

// ---------------------------------------------------------------------------
// Kernel 1: fold BN into weights -> fp16 W' and fp32 bias
// rows 0..511 q, 512..1023 k, 1024..1535 v (into Wqkv/Bqkv), 1536..2047 proj
// ---------------------------------------------------------------------------
__global__ __launch_bounds__(256) void fold_weights_kernel(
    const float* __restrict__ qw, const float* __restrict__ qg, const float* __restrict__ qb,
    const float* __restrict__ qm, const float* __restrict__ qv,
    const float* __restrict__ kw, const float* __restrict__ kg, const float* __restrict__ kb,
    const float* __restrict__ km, const float* __restrict__ kv,
    const float* __restrict__ vw, const float* __restrict__ vg, const float* __restrict__ vb,
    const float* __restrict__ vm, const float* __restrict__ vv,
    const float* __restrict__ pw, const float* __restrict__ pg, const float* __restrict__ pb,
    const float* __restrict__ pm, const float* __restrict__ pv,
    _Float16* __restrict__ Wqkv, float* __restrict__ Bqkv,
    _Float16* __restrict__ Wp,   float* __restrict__ Bp)
{
  const int row = blockIdx.x;          // 0..2047
  const int tid = threadIdx.x;         // 0..255 (handles 2 columns)
  const float *w, *g, *be, *mn, *vr;
  _Float16* wd; float* bd; int o;
  if (row < 512)       { w=qw; g=qg; be=qb; mn=qm; vr=qv; o=row;        wd=Wqkv+(size_t)row*512; bd=Bqkv+row; }
  else if (row < 1024) { w=kw; g=kg; be=kb; mn=km; vr=kv; o=row-512;    wd=Wqkv+(size_t)row*512; bd=Bqkv+row; }
  else if (row < 1536) { w=vw; g=vg; be=vb; mn=vm; vr=vv; o=row-1024;   wd=Wqkv+(size_t)row*512; bd=Bqkv+row; }
  else                 { w=pw; g=pg; be=pb; mn=pm; vr=pv; o=row-1536;   wd=Wp  +(size_t)o  *512; bd=Bp  +o;   }
  const float s = g[o] * rsqrtf(vr[o] + 1e-5f);
  if (tid == 0) *bd = be[o] - mn[o] * s;
  wd[tid]       = (_Float16)(w[(size_t)o*512 + tid]       * s);
  wd[tid + 256] = (_Float16)(w[(size_t)o*512 + tid + 256] * s);
}

// ---------------------------------------------------------------------------
// Kernel 2: quantize x (head_spike) + transpose [tb][c][n] -> [tb][n][c] fp16
// ---------------------------------------------------------------------------
__global__ __launch_bounds__(256) void quantize_x_kernel(
    const float* __restrict__ x, _Float16* __restrict__ xs)
{
  __shared__ float tile[32][33];
  const int tx = threadIdx.x & 31;
  const int ty = threadIdx.x >> 5;    // 0..7
  const int c0 = blockIdx.x * 32;
  const int n0 = blockIdx.y * 32;
  const int tb = blockIdx.z;
  const float* src = x + (size_t)tb * C_DIM * N_POS;
#pragma unroll
  for (int i = 0; i < 4; ++i) {
    const int c = ty + i * 8;
    tile[c][tx] = qspike(src[(size_t)(c0 + c) * N_POS + n0 + tx]);
  }
  __syncthreads();
  _Float16* dst = xs + (size_t)tb * N_POS * C_DIM;
#pragma unroll
  for (int i = 0; i < 4; ++i) {
    const int n = ty + i * 8;
    dst[(size_t)(n0 + n) * C_DIM + c0 + tx] = (_Float16)tile[tx][n];
  }
}

// ---------------------------------------------------------------------------
// Kernel 3: QKV GEMM  [32768,512] x [1536,512]^T, fused bias + q_if, fp16 out
// WG 256 thr = 8 waves, tile 128x128x32, double-buffered LDS, 8 wmma/wave/step
// ---------------------------------------------------------------------------
#define BM  128
#define BN  128
#define BK  32
#define LDT 40    // LDS tile stride (halves): 80B rows, 16B aligned, no bank conflicts
#define LDC 136   // epilogue staging stride

__global__ __launch_bounds__(256) void gemm_qkv_kernel(
    const _Float16* __restrict__ A,    // [32768][512]
    const _Float16* __restrict__ W,    // [1536][512] (o-major, BN folded)
    const float*    __restrict__ bias, // [1536]
    _Float16*       __restrict__ Out)  // [32768][1536] quantized spikes
{
  __shared__ __align__(16) _Float16 smem[2*BM*LDT + 2*BN*LDT];
  _Float16* sA0 = smem;
  _Float16* sA1 = smem +     BM*LDT;
  _Float16* sB0 = smem + 2 * BM*LDT;
  _Float16* sB1 = smem + 2 * BM*LDT + BN*LDT;
  _Float16* sC  = smem;                      // epilogue alias (17408 <= 20480 halves)

  const int tid  = threadIdx.x;
  const int lane = tid & 31;
  const int wave = tid >> 5;
  const int wm   = (wave >> 1) * 32;
  const int wn   = (wave & 1)  * 64;
  const int mtile = blockIdx.x, ntile = blockIdx.y;

  const _Float16* gA = A + (size_t)mtile * BM * 512;
  const _Float16* gB = W + (size_t)ntile * BN * 512;
  const int lrow = tid >> 1;
  const int lcol = (tid & 1) * 16;
  const _Float16* pa = gA + (size_t)lrow * 512 + lcol;
  const _Float16* pb = gB + (size_t)lrow * 512 + lcol;

  v8h ra0 = *(const v8h*)pa, ra1 = *(const v8h*)(pa + 8);
  v8h rb0 = *(const v8h*)pb, rb1 = *(const v8h*)(pb + 8);
  { _Float16* d = sA0 + lrow*LDT + lcol; *(v8h*)d = ra0; *(v8h*)(d+8) = ra1; }
  { _Float16* d = sB0 + lrow*LDT + lcol; *(v8h*)d = rb0; *(v8h*)(d+8) = rb1; }
  __syncthreads();

  const v8f vzero = {};
  v8f acc[2][4];
#pragma unroll
  for (int i = 0; i < 2; ++i)
#pragma unroll
    for (int j = 0; j < 4; ++j) acc[i][j] = vzero;

  for (int ks = 0; ks < 16; ++ks) {
    _Float16* curA = (ks & 1) ? sA1 : sA0;
    _Float16* curB = (ks & 1) ? sB1 : sB0;
    const bool more = (ks < 15);
    if (more) {
      const _Float16* na = pa + (ks + 1) * BK;
      const _Float16* nb = pb + (ks + 1) * BK;
      ra0 = *(const v8h*)na; ra1 = *(const v8h*)(na + 8);
      rb0 = *(const v8h*)nb; rb1 = *(const v8h*)(nb + 8);
      if (ks < 14) {                        // global_prefetch_b8 hints two steps ahead
        __builtin_prefetch(pa + (ks + 2) * BK, 0, 0);
        __builtin_prefetch(pb + (ks + 2) * BK, 0, 0);
      }
    }
    v16h af[2], bf[4];
#pragma unroll
    for (int i = 0; i < 2; ++i) af[i] = load_frag16(curA + (wm + i*16) * LDT, LDT);
#pragma unroll
    for (int j = 0; j < 4; ++j) bf[j] = load_frag16(curB + (wn + j*16) * LDT, LDT);
#pragma unroll
    for (int i = 0; i < 2; ++i)
#pragma unroll
      for (int j = 0; j < 4; ++j)
        acc[i][j] = wmma_f16(af[i], bf[j], acc[i][j]);
    if (more) {
      _Float16* nxA = (ks & 1) ? sA0 : sA1;
      _Float16* nxB = (ks & 1) ? sB0 : sB1;
      _Float16* da = nxA + lrow*LDT + lcol; *(v8h*)da = ra0; *(v8h*)(da+8) = ra1;
      _Float16* db = nxB + lrow*LDT + lcol; *(v8h*)db = rb0; *(v8h*)(db+8) = rb1;
    }
    __syncthreads();
  }

  // epilogue: bias + q_if + fp16, staged through LDS for coalesced stores
  const int r  = lane & 15;
  const int mo = (lane >> 4) << 3;
#pragma unroll
  for (int i = 0; i < 2; ++i)
#pragma unroll
    for (int j = 0; j < 4; ++j) {
      const int col = wn + j*16 + r;
      const float bb = bias[ntile*BN + col];
#pragma unroll
      for (int e = 0; e < 8; ++e)
        sC[(wm + i*16 + mo + e) * LDC + col] = (_Float16)qspike(acc[i][j][e] + bb);
    }
  __syncthreads();
  {
    const int row = tid >> 1;
    const int cs  = (tid & 1) * 64;
    _Float16* dst = Out + ((size_t)(mtile*BM + row)) * QKV_LD + ntile*BN + cs;
    const _Float16* src = sC + row*LDC + cs;
#pragma unroll
    for (int i = 0; i < 8; ++i) *(v8h*)(dst + i*8) = *(const v8h*)(src + i*8);
  }
}

// ---------------------------------------------------------------------------
// Kernel 4: attention per (tb, head): S = 0.25*K^T V (64x64), O = q_if(Q S)/8
// K/V streamed in 64-row transposed chunks; Q fragments read from global.
// Output written [tb][n][h*64+d]: the reference transpose(3,4) cancels here.
// ---------------------------------------------------------------------------
#define KTC 72
#define QSS 72

__global__ __launch_bounds__(256) void attn_kernel(
    const _Float16* __restrict__ QKV,  // [128*256][1536]  (q|k|v)
    _Float16*       __restrict__ Oh)   // [128*256][512]
{
  __shared__ __align__(16) _Float16 smem[64*KTC + 64*KTC + 64*QSS + 256*QSS];
  _Float16* sKt = smem;                 // [64 d1][KTC] transposed K chunk
  _Float16* sVt = sKt + 64*KTC;         // [64 d2][KTC] transposed V chunk
  _Float16* sS  = sVt + 64*KTC;         // [64 d2][QSS] = (0.25*K^T V)^T
  _Float16* sO  = sS  + 64*QSS;         // [256 n][QSS]

  const int tid  = threadIdx.x;
  const int lane = tid & 31;
  const int wave = tid >> 5;
  const int tb   = blockIdx.x >> 3;
  const int h    = blockIdx.x & 7;
  const _Float16* base = QKV + (size_t)tb * N_POS * QKV_LD;
  const int seg = tid & 3;              // 16-half segment of the head dim
  const int nl  = tid >> 2;             // 0..63 local row

  // ---- S = K^T V accumulated over 4 chunks of 64 positions ----
  const v8f vzero = {};
  v8f accS[2] = { vzero, vzero };
  for (int cc = 0; cc < 4; ++cc) {
    const int n = cc*64 + nl;
    const _Float16* kp = base + (size_t)n*QKV_LD +     C_DIM + h*HD + seg*16;
    const _Float16* vp = base + (size_t)n*QKV_LD + 2 * C_DIM + h*HD + seg*16;
    v8h k0 = *(const v8h*)kp, k1 = *(const v8h*)(kp + 8);
    v8h v0 = *(const v8h*)vp, v1 = *(const v8h*)(vp + 8);
#pragma unroll
    for (int i = 0; i < 8; ++i) {
      sKt[(seg*16 + i    ) * KTC + nl] = k0[i];
      sKt[(seg*16 + 8 + i) * KTC + nl] = k1[i];
      sVt[(seg*16 + i    ) * KTC + nl] = v0[i];
      sVt[(seg*16 + 8 + i) * KTC + nl] = v1[i];
    }
    __syncthreads();
#pragma unroll
    for (int rep = 0; rep < 2; ++rep) {
      const int t  = wave + rep*8;      // 16 tiles over (d1,d2)
      const int mi = t >> 2, ni = t & 3;
#pragma unroll
      for (int k = 0; k < 2; ++k) {
        v16h a = load_frag16(sKt + (mi*16) * KTC + k*32, KTC);
        v16h b = load_frag16(sVt + (ni*16) * KTC + k*32, KTC);
        accS[rep] = wmma_f16(a, b, accS[rep]);
      }
    }
    __syncthreads();
  }
  {
    const int r  = lane & 15;
    const int mo = (lane >> 4) << 3;
#pragma unroll
    for (int rep = 0; rep < 2; ++rep) {
      const int t  = wave + rep*8;
      const int mi = t >> 2, ni = t & 3;
#pragma unroll
      for (int e = 0; e < 8; ++e)       // store S transposed: sS[d2][d1]
        sS[(ni*16 + r) * QSS + mi*16 + mo + e] = (_Float16)(accS[rep][e] * 0.25f);
    }
  }
  __syncthreads();

  // ---- O = q_if(Q S)/8 : 16 n-blocks x 4 d2-blocks, K=64 ----
  {
    const int r  = lane & 15;
    const int mo = (lane >> 4) << 3;
#pragma unroll
    for (int mg = 0; mg < 2; ++mg) {
      const int mi = wave*2 + mg;       // 0..15
      const _Float16* qbase = base + (size_t)(mi*16) * QKV_LD + h*HD;
      v16h a0 = load_frag16(qbase,      QKV_LD);   // Q frags straight from L2/HBM
      v16h a1 = load_frag16(qbase + 32, QKV_LD);
#pragma unroll
      for (int ni = 0; ni < 4; ++ni) {
        v16h b0 = load_frag16(sS + (ni*16) * QSS,      QSS);
        v16h b1 = load_frag16(sS + (ni*16) * QSS + 32, QSS);
        v8f acc = vzero;
        acc = wmma_f16(a0, b0, acc);
        acc = wmma_f16(a1, b1, acc);
#pragma unroll
        for (int e = 0; e < 8; ++e)
          sO[(mi*16 + mo + e) * QSS + ni*16 + r] = (_Float16)qspike(acc[e]);
      }
    }
  }
  __syncthreads();

  // ---- coalesced head-slice store: Oh[tb*256+n][h*64 + d] ----
#pragma unroll
  for (int pass = 0; pass < 4; ++pass) {
    const int n = pass*64 + nl;
    const _Float16* src = sO + n*QSS + seg*16;
    _Float16* dst = Oh + ((size_t)(tb*N_POS + n)) * C_DIM + h*HD + seg*16;
    *(v8h*)dst       = *(const v8h*)src;
    *(v8h*)(dst + 8) = *(const v8h*)(src + 8);
  }
}

// ---------------------------------------------------------------------------
// Kernel 5: proj GEMM [32768,512] x [512,512]^T + bias, fp32 out transposed
// to [tb][channel][n]. Fragment rows are contiguous in n -> two float4 stores
// per fragment per lane, no staging needed.
// ---------------------------------------------------------------------------
__global__ __launch_bounds__(256) void gemm_proj_kernel(
    const _Float16* __restrict__ A,    // [32768][512] (attn spikes)
    const _Float16* __restrict__ W,    // [512][512]
    const float*    __restrict__ bias, // [512]
    float*          __restrict__ Out)  // [128][512][256]
{
  __shared__ __align__(16) _Float16 smem[2*BM*LDT + 2*BN*LDT];
  _Float16* sA0 = smem;
  _Float16* sA1 = smem +     BM*LDT;
  _Float16* sB0 = smem + 2 * BM*LDT;
  _Float16* sB1 = smem + 2 * BM*LDT + BN*LDT;

  const int tid  = threadIdx.x;
  const int lane = tid & 31;
  const int wave = tid >> 5;
  const int wm   = (wave >> 1) * 32;
  const int wn   = (wave & 1)  * 64;
  const int mtile = blockIdx.x, ntile = blockIdx.y;

  const _Float16* gA = A + (size_t)mtile * BM * 512;
  const _Float16* gB = W + (size_t)ntile * BN * 512;
  const int lrow = tid >> 1;
  const int lcol = (tid & 1) * 16;
  const _Float16* pa = gA + (size_t)lrow * 512 + lcol;
  const _Float16* pb = gB + (size_t)lrow * 512 + lcol;

  v8h ra0 = *(const v8h*)pa, ra1 = *(const v8h*)(pa + 8);
  v8h rb0 = *(const v8h*)pb, rb1 = *(const v8h*)(pb + 8);
  { _Float16* d = sA0 + lrow*LDT + lcol; *(v8h*)d = ra0; *(v8h*)(d+8) = ra1; }
  { _Float16* d = sB0 + lrow*LDT + lcol; *(v8h*)d = rb0; *(v8h*)(d+8) = rb1; }
  __syncthreads();

  const v8f vzero = {};
  v8f acc[2][4];
#pragma unroll
  for (int i = 0; i < 2; ++i)
#pragma unroll
    for (int j = 0; j < 4; ++j) acc[i][j] = vzero;

  for (int ks = 0; ks < 16; ++ks) {
    _Float16* curA = (ks & 1) ? sA1 : sA0;
    _Float16* curB = (ks & 1) ? sB1 : sB0;
    const bool more = (ks < 15);
    if (more) {
      const _Float16* na = pa + (ks + 1) * BK;
      const _Float16* nb = pb + (ks + 1) * BK;
      ra0 = *(const v8h*)na; ra1 = *(const v8h*)(na + 8);
      rb0 = *(const v8h*)nb; rb1 = *(const v8h*)(nb + 8);
      if (ks < 14) {
        __builtin_prefetch(pa + (ks + 2) * BK, 0, 0);
        __builtin_prefetch(pb + (ks + 2) * BK, 0, 0);
      }
    }
    v16h af[2], bf[4];
#pragma unroll
    for (int i = 0; i < 2; ++i) af[i] = load_frag16(curA + (wm + i*16) * LDT, LDT);
#pragma unroll
    for (int j = 0; j < 4; ++j) bf[j] = load_frag16(curB + (wn + j*16) * LDT, LDT);
#pragma unroll
    for (int i = 0; i < 2; ++i)
#pragma unroll
      for (int j = 0; j < 4; ++j)
        acc[i][j] = wmma_f16(af[i], bf[j], acc[i][j]);
    if (more) {
      _Float16* nxA = (ks & 1) ? sA0 : sA1;
      _Float16* nxB = (ks & 1) ? sB0 : sB1;
      _Float16* da = nxA + lrow*LDT + lcol; *(v8h*)da = ra0; *(v8h*)(da+8) = ra1;
      _Float16* db = nxB + lrow*LDT + lcol; *(v8h*)db = rb0; *(v8h*)(db+8) = rb1;
    }
    __syncthreads();
  }

  // epilogue: bias + transposed store, contiguous in n per lane
  const int r  = lane & 15;
  const int mo = (lane >> 4) << 3;
#pragma unroll
  for (int i = 0; i < 2; ++i) {
    const int mglob = mtile*BM + wm + i*16 + mo;   // global position index base
    const int tb = mglob >> 8;
    const int np = mglob & 255;
#pragma unroll
    for (int j = 0; j < 4; ++j) {
      const int ch = ntile*BN + wn + j*16 + r;
      const float bb = bias[ch];
      float* dst = Out + ((size_t)tb * C_DIM + ch) * N_POS + np;
      v4f s0, s1;
#pragma unroll
      for (int e = 0; e < 4; ++e) { s0[e] = acc[i][j][e] + bb; s1[e] = acc[i][j][e + 4] + bb; }
      *(v4f*)dst       = s0;
      *(v4f*)(dst + 4) = s1;
    }
  }
}

// ---------------------------------------------------------------------------
// Host launch
// ---------------------------------------------------------------------------
extern "C" void kernel_launch(void* const* d_in, const int* in_sizes, int n_in,
                              void* d_out, int out_size, void* d_ws, size_t ws_size,
                              hipStream_t stream)
{
  (void)in_sizes; (void)n_in; (void)out_size; (void)ws_size;

  const float* x  = (const float*)d_in[0];
  const float* qw = (const float*)d_in[1];
  const float* qg = (const float*)d_in[2];
  const float* qb = (const float*)d_in[3];
  const float* qm = (const float*)d_in[4];
  const float* qv = (const float*)d_in[5];
  const float* kw = (const float*)d_in[6];
  const float* kg = (const float*)d_in[7];
  const float* kb = (const float*)d_in[8];
  const float* km = (const float*)d_in[9];
  const float* kv = (const float*)d_in[10];
  const float* vw = (const float*)d_in[11];
  const float* vg = (const float*)d_in[12];
  const float* vb = (const float*)d_in[13];
  const float* vm = (const float*)d_in[14];
  const float* vv = (const float*)d_in[15];
  const float* pw = (const float*)d_in[16];
  const float* pg = (const float*)d_in[17];
  const float* pb = (const float*)d_in[18];
  const float* pm = (const float*)d_in[19];
  const float* pv = (const float*)d_in[20];

  char* ws = (char*)d_ws;
  size_t off = 0;
  auto carve = [&](size_t bytes) -> void* {
    void* p = ws + off;
    off += (bytes + 255) & ~(size_t)255;
    return p;
  };
  _Float16* xs   = (_Float16*)carve((size_t)TB_TOTAL * N_POS * C_DIM * 2); //  33.5 MB
  _Float16* wqkv = (_Float16*)carve((size_t)QKV_LD * C_DIM * 2);           //   1.6 MB
  float*    bqkv = (float*)   carve((size_t)QKV_LD * 4);
  _Float16* wp   = (_Float16*)carve((size_t)C_DIM * C_DIM * 2);
  float*    bp   = (float*)   carve((size_t)C_DIM * 4);
  _Float16* qkv  = (_Float16*)carve((size_t)TB_TOTAL * N_POS * QKV_LD * 2);// 100.7 MB
  _Float16* oh   = (_Float16*)carve((size_t)TB_TOTAL * N_POS * C_DIM * 2); //  33.5 MB

  fold_weights_kernel<<<2048, 256, 0, stream>>>(
      qw, qg, qb, qm, qv, kw, kg, kb, km, kv, vw, vg, vb, vm, vv,
      pw, pg, pb, pm, pv, wqkv, bqkv, wp, bp);

  quantize_x_kernel<<<dim3(C_DIM / 32, N_POS / 32, TB_TOTAL), 256, 0, stream>>>(x, xs);

  gemm_qkv_kernel<<<dim3((TB_TOTAL * N_POS) / BM, QKV_LD / BN), 256, 0, stream>>>(
      xs, wqkv, bqkv, qkv);

  attn_kernel<<<TB_TOTAL * HEADS, 256, 0, stream>>>(qkv, oh);

  gemm_proj_kernel<<<dim3((TB_TOTAL * N_POS) / BM, C_DIM / BN), 256, 0, stream>>>(
      oh, wp, bp, (float*)d_out);
}
